// LSTM2_30399778521541
// MI455X (gfx1250) — matrix operand order repaired and checked by reading, not compiled, not run on hardware
//
#include <hip/hip_runtime.h>
#include <math.h>

// ---------- types ----------
typedef __attribute__((ext_vector_type(16))) __bf16 v16bf;
typedef __attribute__((ext_vector_type(8)))  float  v8f;
typedef int v4i __attribute__((vector_size(16)));

struct __attribute__((aligned(16))) U4 { unsigned x, y, z, w; };

#define LDSA_STRIDE 40   // 32 bf16 + 8 pad (80B rows -> bank spread, 16B aligned)
#define LDSB_STRIDE 40

#define TSTEPS 512
#define BATCH  128
#define DIN    512
#define CSZ    1024
#define GSZ    4096

// ---------- optional gfx1250 async global->LDS staging ----------
#if defined(__has_builtin)
#if __has_builtin(__builtin_amdgcn_global_load_async_to_lds_b128)
#define HAVE_ASYNC_LDS 1
#endif
#endif
#ifndef HAVE_ASYNC_LDS
#define HAVE_ASYNC_LDS 0
#endif

#if HAVE_ASYNC_LDS
__device__ __forceinline__ void async_ld_b128(const void* g, void* l) {
  __builtin_amdgcn_global_load_async_to_lds_b128(
      (__attribute__((address_space(1))) v4i*)g,
      (__attribute__((address_space(3))) v4i*)l, 0, 0);
}
__device__ __forceinline__ void wait_async0() {
#if __has_builtin(__builtin_amdgcn_s_wait_asynccnt)
  __builtin_amdgcn_s_wait_asynccnt(0);
#else
  asm volatile("s_wait_asynccnt 0" ::: "memory");
#endif
}
#endif

// ---------- helpers ----------
// Let the backend pick native bf16 converts (packed v_cvt when available).
__device__ __forceinline__ unsigned short f2bf(float f) {
  union { __bf16 b; unsigned short u; } v;
  v.b = (__bf16)f;
  return v.u;
}
__device__ __forceinline__ unsigned pack_bf2(float lo, float hi) {
  union { __bf16 b[2]; unsigned u; } v;
  v.b[0] = (__bf16)lo;
  v.b[1] = (__bf16)hi;
  return v.u;
}
__device__ __forceinline__ float bf2f(unsigned short h) {
  union { float f; unsigned u; } v; v.u = ((unsigned)h) << 16;
  return v.f;
}
__device__ __forceinline__ float fsig(float x) { return 1.f / (1.f + __expf(-x)); }
__device__ __forceinline__ float ftanh(float x) {
  float cx = fminf(fmaxf(x, -15.f), 15.f);   // avoid inf/inf
  float e = __expf(2.f * cx);                // single v_exp_f32 (TRANS)
  return (e - 1.f) / (e + 1.f);
}

__device__ __forceinline__ U4 ldg_cvt_bf16(const float* __restrict__ s) {
  U4 t;
  t.x = pack_bf2(s[0], s[1]);
  t.y = pack_bf2(s[2], s[3]);
  t.z = pack_bf2(s[4], s[5]);
  t.w = pack_bf2(s[6], s[7]);
  return t;
}

// Build a 16x32 bf16 A-fragment (or 32x16 B-fragment) from two 16B spans.
__device__ __forceinline__ v16bf ld_frag(const unsigned short* p0, const unsigned short* p1) {
  v16bf r;
  unsigned* rp = (unsigned*)&r;
  const unsigned* a = (const unsigned*)p0;
  const unsigned* b = (const unsigned*)p1;
#pragma unroll
  for (int i = 0; i < 4; ++i) { rp[i] = a[i]; rp[4 + i] = b[i]; }
  return r;
}

// ---------- prep: weights fp32->bf16, fused bias, zero state ----------
__global__ void prep_kernel(const float* __restrict__ w_ih, const float* __restrict__ b_ih,
                            const float* __restrict__ w_hh, const float* __restrict__ b_hh,
                            const float* __restrict__ w1,   const float* __restrict__ w2,
                            unsigned short* __restrict__ wih_bf, unsigned short* __restrict__ whh_bf,
                            unsigned short* __restrict__ w1_bf,  unsigned short* __restrict__ w2_bf,
                            float* __restrict__ biasg, unsigned short* __restrict__ hbuf,
                            float* __restrict__ cstate) {
  int tid = blockIdx.x * blockDim.x + threadIdx.x;
  int n   = gridDim.x * blockDim.x;
  for (int i = tid; i < GSZ * DIN; i += n) wih_bf[i] = f2bf(w_ih[i]);
  for (int i = tid; i < GSZ * CSZ; i += n) whh_bf[i] = f2bf(w_hh[i]);
  for (int i = tid; i < CSZ * CSZ; i += n) w1_bf[i] = f2bf(w1[i]);
  for (int i = tid; i < CSZ * CSZ; i += n) w2_bf[i] = f2bf(w2[i]);
  for (int i = tid; i < GSZ; i += n)       biasg[i] = b_ih[i] + b_hh[i];
  for (int i = tid; i < BATCH * CSZ; i += n) { hbuf[i] = 0; cstate[i] = 0.f; }
}

// ---------- fused gates GEMM + LSTM cell ----------
// Grid: 64 WGs x 16 hidden cols (all 4 gates). 8 waves: (gate, mhalf); each wave
// owns 4 M-tiles x 1 N-tile. Ping-pong LDS staging, K=32/iter, 48 iters.
__global__ __launch_bounds__(256) void gates_cell_kernel(
    const float* __restrict__ xt,              // [128][512] fp32 frame
    const unsigned short* __restrict__ hprev,  // [128][1024] bf16
    const unsigned short* __restrict__ wih,    // [4096][512] bf16 row-major = B^T
    const unsigned short* __restrict__ whh,    // [4096][1024] bf16
    const float* __restrict__ biasg,           // [4096] (b_ih + b_hh)
    float* __restrict__ cstate,                // [128][1024] fp32
    unsigned short* __restrict__ hmid)         // [128][1024] bf16
{
  __shared__ unsigned short ldsA[2][BATCH * LDSA_STRIDE];
  __shared__ unsigned short ldsB[2][64 * LDSB_STRIDE];
  __shared__ float gLds[4 * BATCH * 16];

  const int tid   = threadIdx.x;
  const int nh0   = blockIdx.x * 16;
  const int wid   = tid >> 5;
  const int lane  = tid & 31;
  const int gate  = wid >> 1;
  const int mhalf = wid & 1;
  const int hi    = lane >> 4;
  const int l15   = lane & 15;

  // per-thread staging coordinates (two 16B A-chunks, one 16B B-chunk)
  const int rowA0 = tid >> 2, segA = tid & 3;
  const int rowA1 = rowA0 + 64;
  const int lB    = tid >> 2, segB = tid & 3;
  const int growB = (lB >> 4) * CSZ + nh0 + (lB & 15);

  v8f acc[4];
#pragma unroll
  for (int t = 0; t < 4; ++t) {
#pragma unroll
    for (int e = 0; e < 8; ++e) acc[t][e] = 0.f;
  }

  U4 aReg[2]; U4 bReg; bool pendIsX = false;

  auto issue_stage = [&](int kk, int buf) {
    const int k0 = kk * 32;
    if (k0 < DIN) {
      pendIsX = true;
      aReg[0] = ldg_cvt_bf16(xt + rowA0 * DIN + k0 + segA * 8);
      aReg[1] = ldg_cvt_bf16(xt + rowA1 * DIN + k0 + segA * 8);
#if HAVE_ASYNC_LDS
      async_ld_b128(&wih[growB * DIN + k0 + segB * 8],
                    &ldsB[buf][lB * LDSB_STRIDE + segB * 8]);
#else
      bReg = *(const U4*)&wih[growB * DIN + k0 + segB * 8];
#endif
    } else {
      pendIsX = false;
      const int kh = k0 - DIN;
#if HAVE_ASYNC_LDS
      async_ld_b128(&hprev[rowA0 * CSZ + kh + segA * 8],
                    &ldsA[buf][rowA0 * LDSA_STRIDE + segA * 8]);
      async_ld_b128(&hprev[rowA1 * CSZ + kh + segA * 8],
                    &ldsA[buf][rowA1 * LDSA_STRIDE + segA * 8]);
      async_ld_b128(&whh[growB * CSZ + kh + segB * 8],
                    &ldsB[buf][lB * LDSB_STRIDE + segB * 8]);
#else
      aReg[0] = *(const U4*)&hprev[rowA0 * CSZ + kh + segA * 8];
      aReg[1] = *(const U4*)&hprev[rowA1 * CSZ + kh + segA * 8];
      bReg    = *(const U4*)&whh[growB * CSZ + kh + segB * 8];
#endif
    }
  };

  auto complete_stage = [&](int buf) {
    if (pendIsX) {
      *(U4*)&ldsA[buf][rowA0 * LDSA_STRIDE + segA * 8] = aReg[0];
      *(U4*)&ldsA[buf][rowA1 * LDSA_STRIDE + segA * 8] = aReg[1];
#if !HAVE_ASYNC_LDS
      *(U4*)&ldsB[buf][lB * LDSB_STRIDE + segB * 8] = bReg;
#endif
    } else {
#if !HAVE_ASYNC_LDS
      *(U4*)&ldsA[buf][rowA0 * LDSA_STRIDE + segA * 8] = aReg[0];
      *(U4*)&ldsA[buf][rowA1 * LDSA_STRIDE + segA * 8] = aReg[1];
      *(U4*)&ldsB[buf][lB * LDSB_STRIDE + segB * 8] = bReg;
#endif
    }
#if HAVE_ASYNC_LDS
    wait_async0();
#endif
    __syncthreads();
  };

  issue_stage(0, 0);
  complete_stage(0);

  for (int kk = 0; kk < 48; ++kk) {
    const int cur = kk & 1;
    // hoist ALL fragment loads so dscnt waits can be pipelined against WMMAs
    const unsigned short* bp = &ldsB[cur][(gate * 16 + l15) * LDSB_STRIDE + hi * 8];
    v16bf bfrag = ld_frag(bp, bp + 16);
    v16bf afrag[4];
#pragma unroll
    for (int t = 0; t < 4; ++t) {
      const unsigned short* ap = &ldsA[cur][((mhalf * 4 + t) * 16 + l15) * LDSA_STRIDE];
      afrag[t] = ld_frag(ap + hi * 8, ap + 16 + hi * 8);
    }
    // overlap: issue next tile's staging while WMMAs run
    if (kk + 1 < 48) issue_stage(kk + 1, cur ^ 1);
#pragma unroll
    for (int t = 0; t < 4; ++t)
      acc[t] = __builtin_amdgcn_wmma_f32_16x16x32_bf16(
          false, afrag[t], false, bfrag, (short)0, acc[t], false, false);
    if (kk + 1 < 48) complete_stage(cur ^ 1);
  }

  // scatter accumulators to LDS gates buffer (C layout: vgpr e -> M=e / M=8+e)
#pragma unroll
  for (int t = 0; t < 4; ++t) {
    int rowb = (mhalf * 4 + t) * 16 + hi * 8;
#pragma unroll
    for (int e = 0; e < 8; ++e)
      gLds[(gate * BATCH + rowb + e) * 16 + l15] = acc[t][e];
  }
  __syncthreads();

  // fused LSTM cell elementwise over 128x16 values
#pragma unroll
  for (int i = 0; i < 8; ++i) {
    int id = tid + 256 * i;           // 0..2047
    int row = id >> 4, col = id & 15;
    int hcol = nh0 + col;
    float iv = gLds[(0 * BATCH + row) * 16 + col] + biasg[hcol];
    float fv = gLds[(1 * BATCH + row) * 16 + col] + biasg[CSZ + hcol];
    float gv = gLds[(2 * BATCH + row) * 16 + col] + biasg[2 * CSZ + hcol];
    float ov = gLds[(3 * BATCH + row) * 16 + col] + biasg[3 * CSZ + hcol];
    float cOld = cstate[row * CSZ + hcol];
    float cN = fsig(fv) * cOld + fsig(iv) * ftanh(gv);
    float hN = fsig(ov) * ftanh(cN);
    cstate[row * CSZ + hcol] = cN;
    hmid[row * CSZ + hcol] = f2bf(hN);
  }
}

// ---------- MLP layer: out = act @ w^T + b (optional relu) ----------
// Grid: 64 WGs x 16 cols; 8 waves = 8 M-tiles; K=1024, ping-pong staging.
__global__ __launch_bounds__(256) void mlp_kernel(
    const unsigned short* __restrict__ act,  // [128][1024] bf16
    const unsigned short* __restrict__ w,    // [1024][1024] bf16 row-major
    const float* __restrict__ bias,          // [1024]
    unsigned short* __restrict__ out,        // [128][1024] bf16
    int do_relu)
{
  __shared__ unsigned short ldsA[2][BATCH * LDSA_STRIDE];
  __shared__ unsigned short ldsB[2][16 * LDSB_STRIDE];

  const int tid  = threadIdx.x;
  const int nh0  = blockIdx.x * 16;
  const int wid  = tid >> 5;
  const int lane = tid & 31;
  const int hi   = lane >> 4;
  const int l15  = lane & 15;

  const int rowA0 = tid >> 2, segA = tid & 3;
  const int rowA1 = rowA0 + 64;
  const int lB    = tid >> 2, segB = tid & 3;   // valid for tid < 64

  v8f acc;
#pragma unroll
  for (int e = 0; e < 8; ++e) acc[e] = 0.f;

  U4 aReg[2]; U4 bReg;

  auto issue_stage = [&](int kk, int buf) {
    const int k0 = kk * 32;
#if HAVE_ASYNC_LDS
    async_ld_b128(&act[rowA0 * CSZ + k0 + segA * 8],
                  &ldsA[buf][rowA0 * LDSA_STRIDE + segA * 8]);
    async_ld_b128(&act[rowA1 * CSZ + k0 + segA * 8],
                  &ldsA[buf][rowA1 * LDSA_STRIDE + segA * 8]);
    if (tid < 64)
      async_ld_b128(&w[(nh0 + lB) * CSZ + k0 + segB * 8],
                    &ldsB[buf][lB * LDSB_STRIDE + segB * 8]);
#else
    aReg[0] = *(const U4*)&act[rowA0 * CSZ + k0 + segA * 8];
    aReg[1] = *(const U4*)&act[rowA1 * CSZ + k0 + segA * 8];
    if (tid < 64) bReg = *(const U4*)&w[(nh0 + lB) * CSZ + k0 + segB * 8];
#endif
  };

  auto complete_stage = [&](int buf) {
#if HAVE_ASYNC_LDS
    wait_async0();
#else
    *(U4*)&ldsA[buf][rowA0 * LDSA_STRIDE + segA * 8] = aReg[0];
    *(U4*)&ldsA[buf][rowA1 * LDSA_STRIDE + segA * 8] = aReg[1];
    if (tid < 64) *(U4*)&ldsB[buf][lB * LDSB_STRIDE + segB * 8] = bReg;
#endif
    __syncthreads();
  };

  issue_stage(0, 0);
  complete_stage(0);

  for (int kk = 0; kk < 32; ++kk) {
    const int cur = kk & 1;
    const unsigned short* bp = &ldsB[cur][l15 * LDSB_STRIDE + hi * 8];
    v16bf bfrag = ld_frag(bp, bp + 16);
    const unsigned short* ap = &ldsA[cur][(wid * 16 + l15) * LDSA_STRIDE];
    v16bf afrag = ld_frag(ap + hi * 8, ap + 16 + hi * 8);
    if (kk + 1 < 32) issue_stage(kk + 1, cur ^ 1);
    acc = __builtin_amdgcn_wmma_f32_16x16x32_bf16(
        false, afrag, false, bfrag, (short)0, acc, false, false);
    if (kk + 1 < 32) complete_stage(cur ^ 1);
  }

  const int col = nh0 + l15;
  const float b = bias[col];
#pragma unroll
  for (int e = 0; e < 8; ++e) {
    int row = wid * 16 + hi * 8 + e;
    float v = acc[e] + b;
    if (do_relu) v = fmaxf(v, 0.f);
    out[row * CSZ + col] = f2bf(v);
  }
}

// ---------- final projection: out = h @ wo^T + bo  (128x3, tiny) ----------
__global__ void final_kernel(const unsigned short* __restrict__ h,
                             const float* __restrict__ wo, const float* __restrict__ bo,
                             float* __restrict__ out) {
  int tid = threadIdx.x;
  if (tid >= 384) return;
  int row = tid & 127;
  int o   = tid >> 7;   // 0..2
  float s = 0.f;
  for (int k = 0; k < CSZ; ++k)
    s += bf2f(h[row * CSZ + k]) * wo[o * CSZ + k];
  out[row * 3 + o] = s + bo[o];
}

// ---------- launch ----------
extern "C" void kernel_launch(void* const* d_in, const int* in_sizes, int n_in,
                              void* d_out, int out_size, void* d_ws, size_t ws_size,
                              hipStream_t stream) {
  (void)in_sizes; (void)n_in; (void)out_size; (void)ws_size;
  const float* x    = (const float*)d_in[0];
  const float* w_ih = (const float*)d_in[1];
  const float* b_ih = (const float*)d_in[2];
  const float* w_hh = (const float*)d_in[3];
  const float* b_hh = (const float*)d_in[4];
  const float* w1   = (const float*)d_in[5];
  const float* b1   = (const float*)d_in[6];
  const float* w2   = (const float*)d_in[7];
  const float* b2   = (const float*)d_in[8];
  const float* wo   = (const float*)d_in[9];
  const float* bo   = (const float*)d_in[10];

  // workspace carve-up (all offsets keep 16B alignment)
  unsigned short* wih_bf = (unsigned short*)d_ws;
  unsigned short* whh_bf = wih_bf + (size_t)GSZ * DIN;
  unsigned short* w1_bf  = whh_bf + (size_t)GSZ * CSZ;
  unsigned short* w2_bf  = w1_bf + (size_t)CSZ * CSZ;
  unsigned short* hbuf   = w2_bf + (size_t)CSZ * CSZ;
  unsigned short* hmid   = hbuf + (size_t)BATCH * CSZ;
  unsigned short* y1     = hmid + (size_t)BATCH * CSZ;
  float* biasg           = (float*)(y1 + (size_t)BATCH * CSZ);
  float* cstate          = biasg + GSZ;

  prep_kernel<<<512, 256, 0, stream>>>(w_ih, b_ih, w_hh, b_hh, w1, w2,
                                       wih_bf, whh_bf, w1_bf, w2_bf,
                                       biasg, hbuf, cstate);

  for (int t = 0; t < TSTEPS; ++t) {
    const float* xt = x + (size_t)t * BATCH * DIN;
    gates_cell_kernel<<<CSZ / 16, 256, 0, stream>>>(xt, hbuf, wih_bf, whh_bf,
                                                    biasg, cstate, hmid);
    mlp_kernel<<<CSZ / 16, 256, 0, stream>>>(hmid, w1_bf, b1, y1, 1);
    mlp_kernel<<<CSZ / 16, 256, 0, stream>>>(y1, w2_bf, b2, hbuf, 0);
  }

  final_kernel<<<1, 384, 0, stream>>>(hbuf, wo, bo, (float*)d_out);
}
